// LSTM_89000312307738
// MI455X (gfx1250) — matrix operand order repaired and checked by reading, not compiled, not run on hardware
//
#include <hip/hip_runtime.h>

// ============================================================================
// Persistent LSTMP for MI455X (gfx1250, wave32, WMMA).
//  - 64 persistent workgroups x 256 threads (8 wave32). WG w owns hidden units
//    j in [16w, 16w+16) for the whole layer: its 64 gate-weight rows
//    ([Wih;Whh] -> 64x512 bf16 = 64KB) and its projection columns
//    (Whr slice 16x1024 bf16 = 32KB) stay resident in LDS across all T steps.
//  - Cell state c (64x16 fp32) never leaves the WGP.
//  - Per step: gate GEMM (K=512 = x_t||h) via v_wmma_f32_16x16x32_bf16,
//    elementwise LSTM cell in fp32, global ticket-barrier, projection GEMM
//    (split-K over 8 waves), barrier. State s/h exchanged as bf16 through L2.
// ============================================================================

typedef __attribute__((ext_vector_type(16))) __bf16 v16bf;
typedef __attribute__((ext_vector_type(8)))  float  v8f;

namespace {
constexpr int B = 64, T = 1024, P = 256, H = 1024, L = 4;
constexpr int NWG  = 64;
constexpr int NTHR = 256;

// workspace layout (bytes)
constexpr size_t OFF_CNT  = 0;                                  // ticket counter
constexpr size_t OFF_ACTA = 256;
constexpr size_t SZ_ACT   = (size_t)B * T * P * 2;              // bf16 activations
constexpr size_t OFF_ACTB = OFF_ACTA + SZ_ACT;
constexpr size_t SZ_WIH   = (size_t)L * 4 * H * P * 2;
constexpr size_t OFF_WIH  = OFF_ACTB + SZ_ACT;
constexpr size_t OFF_WHH  = OFF_WIH + SZ_WIH;
constexpr size_t SZ_WHR   = (size_t)L * P * H * 2;
constexpr size_t OFF_WHR  = OFF_WHH + SZ_WIH;
constexpr size_t SZ_BS    = (size_t)L * 4 * H * 4;              // fused bias fp32
constexpr size_t OFF_BS   = OFF_WHR + SZ_WHR;
constexpr size_t OFF_HBUF = OFF_BS + SZ_BS;                     // h state bf16 [B,P]
constexpr size_t OFF_SBUF = OFF_HBUF + (size_t)B * P * 2;       // s = o*tanh(c) bf16 [B,H]
} // namespace

__device__ __forceinline__ __bf16 f2bf(float f) {
  union { float f; unsigned u; } in; in.f = f;
  unsigned r = (in.u + 0x7FFFu + ((in.u >> 16) & 1u)) >> 16;   // RNE
  union { unsigned short s; __bf16 b; } out; out.s = (unsigned short)r;
  return out.b;
}
__device__ __forceinline__ float sigm(float x)  { return 1.0f / (1.0f + __expf(-x)); }
__device__ __forceinline__ float tanhf_(float x){ float e = __expf(-2.0f * x); return (1.0f - e) / (1.0f + e); }

union FragBF { v16bf v; uint4 q[2]; };

// grid-wide ticket barrier: monotone counter, no reset race. Counter is zeroed
// by prep() each launch, so the kernel is deterministic across graph replays.
__device__ __forceinline__ void gridbar(unsigned* cnt) {
  __threadfence();
  __syncthreads();
  if (threadIdx.x == 0) {
    unsigned t = __hip_atomic_fetch_add(cnt, 1u, __ATOMIC_ACQ_REL, __HIP_MEMORY_SCOPE_AGENT);
    unsigned target = (t / NWG + 1u) * NWG;
    while (__hip_atomic_load(cnt, __ATOMIC_ACQUIRE, __HIP_MEMORY_SCOPE_AGENT) < target)
      __builtin_amdgcn_s_sleep(8);
  }
  __syncthreads();
  __threadfence();
}

// ---------------------------------------------------------------------------
// prep: fp32 -> bf16 for y / Wih / Whh / Whr, fuse biases, zero the counter.
// ---------------------------------------------------------------------------
__global__ void __launch_bounds__(NTHR) prep_kernel(
    const float* __restrict__ y, const float* __restrict__ Wih,
    const float* __restrict__ Whh, const float* __restrict__ bih,
    const float* __restrict__ bhh, const float* __restrict__ Whr,
    unsigned char* __restrict__ ws) {
  const long long NY   = (long long)B * T * P;
  const long long NWIH = (long long)L * 4 * H * P;
  const long long NWHR = (long long)L * P * H;
  const long long NB   = (long long)L * 4 * H;
  long long i = (long long)blockIdx.x * NTHR + threadIdx.x;
  if (i < NY) { ((__bf16*)(ws + OFF_ACTA))[i] = f2bf(y[i]); return; }
  i -= NY;
  if (i < NWIH) { ((__bf16*)(ws + OFF_WIH))[i] = f2bf(Wih[i]); return; }
  i -= NWIH;
  if (i < NWIH) { ((__bf16*)(ws + OFF_WHH))[i] = f2bf(Whh[i]); return; }
  i -= NWIH;
  if (i < NWHR) { ((__bf16*)(ws + OFF_WHR))[i] = f2bf(Whr[i]); return; }
  i -= NWHR;
  if (i < NB) { ((float*)(ws + OFF_BS))[i] = bih[i] + bhh[i]; return; }
  i -= NB;
  if (i == 0) { ((unsigned*)(ws + OFF_CNT))[0] = 0u; }
}

// ---------------------------------------------------------------------------
// persistent LSTM kernel
// ---------------------------------------------------------------------------
__global__ void __launch_bounds__(NTHR) lstm_persist_kernel(unsigned char* __restrict__ ws) {
  const int w    = blockIdx.x;          // 0..63, owns j in [16w,16w+16)
  const int tid  = threadIdx.x;
  const int lane = tid & 31;
  const int wv   = tid >> 5;            // wave 0..7
  const int r    = lane & 15;
  const int half = lane >> 4;

  __bf16* actA = (__bf16*)(ws + OFF_ACTA);
  __bf16* actB = (__bf16*)(ws + OFF_ACTB);
  const __bf16* wih = (const __bf16*)(ws + OFF_WIH);
  const __bf16* whh = (const __bf16*)(ws + OFF_WHH);
  const __bf16* whr = (const __bf16*)(ws + OFF_WHR);
  const float*  bs  = (const float*)(ws + OFF_BS);
  __bf16*  hbuf = (__bf16*)(ws + OFF_HBUF);
  __bf16*  sbuf = (__bf16*)(ws + OFF_SBUF);
  unsigned* cnt = (unsigned*)(ws + OFF_CNT);

  // LDS: padded row strides so 16-row B-fragment reads hit all 64 banks.
  __shared__ __bf16 Wg[64][520];        // [gate*16+jl][k]  k<256: Wih, k>=256: Whh
  __shared__ __bf16 Wr[16][1032];       // [p_local][j]     projection slice
  __shared__ float  gatesS[64][4][16];  // [b][gate][jl]
  __shared__ float  cS[64][16];         // cell state, never leaves WGP
  __shared__ float  predS[8][32][9];    // split-K partials (padded)

  const int pset = (w >> 2) * 16;       // projection output columns
  const int bset = (w & 3) * 16;        // projection output rows (batch)

  for (int l = 0; l < L; ++l) {
    const __bf16* actin  = (l & 1) ? actB : actA;
    __bf16*       actout = (l & 1) ? actA : actB;

    // ---- load this WG's weight slices into LDS (once per layer) ----
    for (int e = tid; e < 64 * 128; e += NTHR) {          // Wih part (uint = 2 bf16)
      int row = e >> 7, c2 = e & 127;
      int G = (row >> 4) * H + w * 16 + (row & 15);
      ((unsigned*)&Wg[row][0])[c2] =
          ((const unsigned*)(wih + ((size_t)l * 4 * H + G) * P))[c2];
    }
    for (int e = tid; e < 64 * 128; e += NTHR) {          // Whh part
      int row = e >> 7, c2 = e & 127;
      int G = (row >> 4) * H + w * 16 + (row & 15);
      ((unsigned*)&Wg[row][256])[c2] =
          ((const unsigned*)(whh + ((size_t)l * 4 * H + G) * P))[c2];
    }
    for (int e = tid; e < 16 * 512; e += NTHR) {          // Whr slice
      int row = e >> 9, c2 = e & 511;
      ((unsigned*)&Wr[row][0])[c2] =
          ((const unsigned*)(whr + ((size_t)l * P + pset + row) * H))[c2];
    }
    for (int e = tid; e < 128; e += NTHR)                 // zero h state (b=w row)
      ((unsigned*)hbuf)[w * 128 + e] = 0u;
    for (int e = tid; e < 64 * 16; e += NTHR)             // zero cell state
      ((float*)cS)[e] = 0.0f;
    gridbar(cnt);

    for (int t = 0; t < T; ++t) {
      // ---- gate GEMM: D[64 x 64] = [x_t || h] (K=512) x Wg^T, bf16 WMMA ----
#pragma unroll
      for (int tt = 0; tt < 2; ++tt) {
        const int ti = wv + tt * 8;                       // 16 tiles (mi, gate)
        const int mi = ti >> 2, g = ti & 3;
        const int brow = 16 * mi + r;
        const __bf16* xrow = actin + ((size_t)brow * T + t) * P;
        const __bf16* hrow = hbuf + brow * P;
        v8f acc = {};
#pragma unroll
        for (int kc = 0; kc < 16; ++kc) {
          const int kk = kc * 32;
          // A frag (16x32 bf16): lane covers K in {8h..8h+7} U {16+8h..16+8h+7}
          const __bf16* asrc = (kc < 8) ? (xrow + kk + 8 * half)
                                        : (hrow + (kk - 256) + 8 * half);
          FragBF a, b;
          a.q[0] = *(const uint4*)(asrc);
          a.q[1] = *(const uint4*)(asrc + 16);
          // B frag (32x16): lane covers K in {8h..8h+7} U {16+8h..16+8h+7}
          const __bf16* bsrc = &Wg[g * 16 + r][kk + 8 * half];
          b.q[0] = *(const uint4*)(bsrc);
          b.q[1] = *(const uint4*)(bsrc + 16);
          acc = __builtin_amdgcn_wmma_f32_16x16x32_bf16(
              false, a.v, false, b.v, (short)0, acc, false, false);
        }
#pragma unroll
        for (int v = 0; v < 8; ++v)                       // C: M=v+8*half, N=lane&15
          gatesS[16 * mi + v + 8 * half][g][r] = acc[v];
      }
      __syncthreads();

      // ---- elementwise LSTM cell (fp32), emit s = sigmoid(o)*tanh(c) ----
      for (int it = tid; it < 64 * 16; it += NTHR) {
        const int b = it >> 4, jl = it & 15;
        const int jg = w * 16 + jl;
        float xi = gatesS[b][0][jl] + bs[(l * 4 + 0) * H + jg];
        float xf = gatesS[b][1][jl] + bs[(l * 4 + 1) * H + jg];
        float xg = gatesS[b][2][jl] + bs[(l * 4 + 2) * H + jg];
        float xo = gatesS[b][3][jl] + bs[(l * 4 + 3) * H + jg];
        float c = sigm(xf) * cS[b][jl] + sigm(xi) * tanhf_(xg);
        cS[b][jl] = c;
        sbuf[b * H + jg] = f2bf(sigm(xo) * tanhf_(c));
      }
      gridbar(cnt);                                       // all s slices visible

      // ---- projection: h[bset..+16, pset..+16] = s (K=1024) x Whr^T ----
      {
        v8f acc = {};
        const __bf16* srow = sbuf + (size_t)(bset + r) * H + wv * 128;  // split-K
#pragma unroll
        for (int kc = 0; kc < 4; ++kc) {
          FragBF a, b;
          const __bf16* asrc = srow + kc * 32 + 8 * half;
          a.q[0] = *(const uint4*)(asrc);
          a.q[1] = *(const uint4*)(asrc + 16);
          const __bf16* bsrc = &Wr[r][wv * 128 + kc * 32 + 8 * half];
          b.q[0] = *(const uint4*)(bsrc);
          b.q[1] = *(const uint4*)(bsrc + 16);
          acc = __builtin_amdgcn_wmma_f32_16x16x32_bf16(
              false, a.v, false, b.v, (short)0, acc, false, false);
        }
#pragma unroll
        for (int v = 0; v < 8; ++v) predS[wv][lane][v] = acc[v];
      }
      __syncthreads();
      if (wv == 0) {                                      // reduce split-K, emit h
#pragma unroll
        for (int v = 0; v < 8; ++v) {
          float s = 0.0f;
#pragma unroll
          for (int u = 0; u < 8; ++u) s += predS[u][lane][v];
          const int b = bset + v + 8 * half;
          const int p = pset + r;
          const __bf16 hb = f2bf(s);
          hbuf[b * P + p] = hb;
          actout[((size_t)b * T + t) * P + p] = hb;
        }
      }
      gridbar(cnt);                                       // h visible for t+1
    }
  }
}

// ---------------------------------------------------------------------------
// finalize: bf16 layer-3 activations -> fp32 output, dropping t < min_seq_len
// ---------------------------------------------------------------------------
__global__ void __launch_bounds__(NTHR) finalize_kernel(
    const unsigned char* __restrict__ ws, float* __restrict__ out,
    int msl, int n) {
  int i = blockIdx.x * NTHR + threadIdx.x;
  if (i >= n) return;
  const int Tm = T - msl;
  const int p = i & (P - 1);
  const int rem = i >> 8;
  const int t = rem % Tm + msl;
  const int b = rem / Tm;
  const __bf16* act = (const __bf16*)(ws + OFF_ACTA);     // layer 3 output buffer
  out[i] = (float)act[((size_t)b * T + t) * P + p];
}

extern "C" void kernel_launch(void* const* d_in, const int* in_sizes, int n_in,
                              void* d_out, int out_size, void* d_ws, size_t ws_size,
                              hipStream_t stream) {
  const float* y   = (const float*)d_in[0];
  const float* Wih = (const float*)d_in[1];
  const float* Whh = (const float*)d_in[2];
  const float* bih = (const float*)d_in[3];
  const float* bhh = (const float*)d_in[4];
  const float* Whr = (const float*)d_in[5];
  unsigned char* ws = (unsigned char*)d_ws;

  // msl is deterministic from out_size: out = [B, T-msl, P]
  int msl = T - out_size / (B * P);
  if (msl < 0) msl = 0;
  if (msl >= T) msl = T - 1;

  const long long prep_total = (long long)B * T * P + 2LL * L * 4 * H * P +
                               (long long)L * P * H + (long long)L * 4 * H + 256;
  const int prep_blocks = (int)((prep_total + NTHR - 1) / NTHR);

  prep_kernel<<<prep_blocks, NTHR, 0, stream>>>(y, Wih, Whh, bih, bhh, Whr, ws);
  lstm_persist_kernel<<<NWG, NTHR, 0, stream>>>(ws);
  finalize_kernel<<<(out_size + NTHR - 1) / NTHR, NTHR, 0, stream>>>(
      ws, (float*)d_out, msl, out_size);
}
